// SkipGramModel_13735305413412
// MI455X (gfx1250) — matrix operand order repaired and checked by reading, not compiled, not run on hardware
//
#include <hip/hip_runtime.h>
#include <math.h>

// Problem constants (match reference)
#define B_    128
#define L_    80
#define W_    5
#define NEG_  5
#define D_    128
#define BL_   (B_ * L_)
#define PPB_  (770 * NEG_)    // 3850 negative pairs per walk
#define LAP   0.01f
#define RSX_  68              // bf16 staging row stride in dwords (136 ushorts = 272B)

typedef __attribute__((ext_vector_type(16))) __bf16 v16bf;
typedef __attribute__((ext_vector_type(8)))  float  v8f;

// Pack two f32 into one dword of bf16 (round-to-nearest-even).
static __device__ __forceinline__ unsigned pack_bf16(float a, float b) {
  unsigned x = __float_as_uint(a); x += 0x7FFFu + ((x >> 16) & 1u);
  unsigned y = __float_as_uint(b); y += 0x7FFFu + ((y >> 16) & 1u);
  return (x >> 16) | (y & 0xFFFF0000u);
}

// Branchless reproduction of the reference sigmoid LUT lookup:
// table[idx] = sigmoid(-6.01 + 0.01*idx), endpoints forced to 0 / 1.
// v_rcp_f32 (~1 ULP) is ample for a 0.01-granularity LUT.
static __device__ __forceinline__ float lut_sigmoid(float x) {
  x = fminf(fmaxf(x, -6.0f), 6.0f);
  int idx = (int)floorf((x + 6.01f) * 100.0f);
  idx = idx < 0 ? 0 : (idx > 1201 ? 1201 : idx);
  float q = -6.01f + 0.01f * (float)idx;
  float s = __builtin_amdgcn_rcpf(1.0f + __expf(-q));
  s = (idx == 0)    ? 0.0f : s;
  s = (idx == 1201) ? 1.0f : s;
  return s;
}

__global__ void zero_out_kernel(float* __restrict__ p, int n) {
  for (int i = blockIdx.x * blockDim.x + threadIdx.x; i < n;
       i += gridDim.x * blockDim.x)
    p[i] = 0.0f;
}

__global__ __launch_bounds__(256)
void skipgram_grad_kernel(const float* __restrict__ uW,
                          const float* __restrict__ vW,
                          const int*   __restrict__ nodes,
                          const int*   __restrict__ negu,
                          const int*   __restrict__ negv,
                          float* __restrict__ outU,
                          float* __restrict__ outV) {
  // f32 copies (exact gradient math); d-contiguous access -> no padding needed
  __shared__ float ldsU[L_ * D_];
  __shared__ float ldsV[L_ * D_];
  // Overlay region: phase B = bf16 staging (2 x 80 x 136 ushorts, padded to
  // break LDS bank serialization on row-strided fragment loads);
  // phase C+  = grad_u accumulator (80 x 128 f32).
  __shared__ __align__(16) unsigned char ldsXraw[2 * L_ * RSX_ * 4];
  __shared__ float ldsP[L_ * 11];    // band weights: s_pos(j,i)+LAP at [j][i-j+5]

  unsigned*  bfU   = (unsigned*)ldsXraw;
  unsigned*  bfV   = (unsigned*)(ldsXraw + L_ * RSX_ * 4);
  float*     gradU = (float*)ldsXraw;

  const int b    = blockIdx.x;
  const int tid  = threadIdx.x;
  const int wave = tid >> 5;
  const int lane = tid & 31;

  // ---------------- Phase A: gather embeddings; stage f32 + bf16 ----------
  for (int t = tid; t < L_ * (D_ / 4); t += 256) {
    int l = t >> 5;             // D_/4 == 32
    int q = t & 31;
    int node = nodes[b * L_ + l];
    float4 u4 = ((const float4*)(uW + (size_t)node * D_))[q];
    float4 v4 = ((const float4*)(vW + (size_t)node * D_))[q];
    *(float4*)(ldsU + l * D_ + q * 4) = u4;
    *(float4*)(ldsV + l * D_ + q * 4) = v4;
    uint2 ub; ub.x = pack_bf16(u4.x, u4.y); ub.y = pack_bf16(u4.z, u4.w);
    uint2 vb; vb.x = pack_bf16(v4.x, v4.y); vb.y = pack_bf16(v4.z, v4.w);
    *(uint2*)(bfU + l * RSX_ + q * 2) = ub;
    *(uint2*)(bfV + l * RSX_ + q * 2) = vb;
  }
  for (int t = tid; t < L_ * 11; t += 256) ldsP[t] = 0.0f;   // band-invalid = 0
  __syncthreads();

  // ---------------- Phase B: banded 80x80 scores via bf16 WMMA ------------
  // S = E_u * E_v^T ; accumulator lanes fold straight into band weights P.
  {
    const int tr[13] = {0,0,1,1,1,2,2,2,3,3,3,4,4};
    const int tc[13] = {0,1,0,1,2,1,2,3,2,3,4,3,4};
    const int half = lane >> 4;   // 0: lanes 0-15, 1: lanes 16-31
    const int m    = lane & 15;
    for (int t = wave; t < 13; t += 8) {
      const int r = tr[t], c = tc[t];
      const unsigned* arow = bfU + (16 * r + m) * RSX_;  // A: M = m
      const unsigned* brow = bfV + (16 * c + m) * RSX_;  // B: N = m (B = E_v^T)
      v8f acc = {0.f, 0.f, 0.f, 0.f, 0.f, 0.f, 0.f, 0.f};
      #pragma unroll
      for (int kk = 0; kk < 4; ++kk) {     // K = 128 in chunks of 32
        union { v16bf v; uint4 q[2]; } Af, Bf;
        // A 16x32 bf16 layout: lanes<16 hold K {0..7,16..23}, lanes>=16 {8..15,24..31}
        const int ka = 32 * kk + (half ? 8 : 0);      // element index (even)
        Af.q[0] = *(const uint4*)(arow + (ka >> 1));
        Af.q[1] = *(const uint4*)(arow + (ka >> 1) + 8);
        // B 32x16 bf16 layout: lanes<16 hold K {0..7,16..23}, lanes>=16 {8..15,24..31}
        const int kb = 32 * kk + (half ? 8 : 0);
        Bf.q[0] = *(const uint4*)(brow + (kb >> 1));
        Bf.q[1] = *(const uint4*)(brow + (kb >> 1) + 8);
        acc = __builtin_amdgcn_wmma_f32_16x16x32_bf16(
            false, Af.v, false, Bf.v, (short)0, acc, false, false);
      }
      // C layout: VGPR p -> M = p + 8*half, N = m.  Band entries are
      // tile-unique, so each valid (j,i) is written exactly once.
      #pragma unroll
      for (int p = 0; p < 8; ++p) {
        int j = 16 * r + p + 8 * half;
        int i = 16 * c + m;
        int o = i - j + 5;
        if ((unsigned)o < 11u && o != 5)
          ldsP[j * 11 + o] = (1.0f - lut_sigmoid(acc[p])) + LAP;
      }
    }
  }
  __syncthreads();

  // ---------------- Phase C: positive gradients (exact f32) ---------------
  // grad_u[j] = sum_i w(j,i)*v_i - LAP*deg(j)*u_j   (into LDS accumulator)
  // grad_v[j] = sum_i w(i,j)*u_i - LAP*deg(j)*v_j   (atomic into outV)
  for (int t = tid; t < L_ * D_; t += 256) {
    int j = t >> 7, d = t & (D_ - 1);
    float accU = 0.f, accV = 0.f;
    int deg = 0;
    #pragma unroll
    for (int o = 0; o < 11; ++o) {
      int i = j + o - 5;
      if (o == 5 || i < 0 || i >= L_) continue;
      ++deg;
      accU += ldsP[j * 11 + o]        * ldsV[i * D_ + d];
      accV += ldsP[i * 11 + (10 - o)] * ldsU[i * D_ + d];
    }
    accU -= LAP * (float)deg * ldsU[j * D_ + d];
    accV -= LAP * (float)deg * ldsV[j * D_ + d];
    gradU[j * D_ + d] = accU;    // overwrites dead bf16 staging (post-barrier)
    __hip_atomic_fetch_add(&outV[(size_t)(b * L_ + j) * D_ + d], accV,
                           __ATOMIC_RELAXED, __HIP_MEMORY_SCOPE_AGENT);
  }
  __syncthreads();

  // ---------------- Phase D: negative pairs (one wave per pair) -----------
  const int pbase = b * PPB_;
  for (int p = wave; p < PPB_; p += 8) {
    const int gu = negu[pbase + p];
    const int gv = negv[pbase + p];
    const int c  = gu - b * L_;                    // local center row
    const int node = nodes[gv];
    const float* vr = vW + (size_t)node * D_;
    const int d0 = lane * 4;
    if (p + 8 < PPB_) {                            // prefetch next pair's v row
      int ng = negv[pbase + p + 8];
      __builtin_prefetch(vW + (size_t)nodes[ng] * D_ + d0, 0, 3);
    }
    float4 vv = *(const float4*)(vr + d0);
    float4 uu = *(const float4*)(ldsU + c * D_ + d0);
    float part = vv.x * uu.x + vv.y * uu.y + vv.z * uu.z + vv.w * uu.w;
    #pragma unroll
    for (int off = 16; off > 0; off >>= 1) part += __shfl_xor(part, off, 32);
    const float s = -lut_sigmoid(part);            // NEG_WEIGHT == 1.0
    float* gvp = outV + (size_t)gv * D_ + d0;
    __hip_atomic_fetch_add(gvp + 0, s * uu.x, __ATOMIC_RELAXED, __HIP_MEMORY_SCOPE_AGENT);
    __hip_atomic_fetch_add(gvp + 1, s * uu.y, __ATOMIC_RELAXED, __HIP_MEMORY_SCOPE_AGENT);
    __hip_atomic_fetch_add(gvp + 2, s * uu.z, __ATOMIC_RELAXED, __HIP_MEMORY_SCOPE_AGENT);
    __hip_atomic_fetch_add(gvp + 3, s * uu.w, __ATOMIC_RELAXED, __HIP_MEMORY_SCOPE_AGENT);
    float* gup = gradU + c * D_ + d0;
    __hip_atomic_fetch_add(gup + 0, s * vv.x, __ATOMIC_RELAXED, __HIP_MEMORY_SCOPE_WORKGROUP);
    __hip_atomic_fetch_add(gup + 1, s * vv.y, __ATOMIC_RELAXED, __HIP_MEMORY_SCOPE_WORKGROUP);
    __hip_atomic_fetch_add(gup + 2, s * vv.z, __ATOMIC_RELAXED, __HIP_MEMORY_SCOPE_WORKGROUP);
    __hip_atomic_fetch_add(gup + 3, s * vv.w, __ATOMIC_RELAXED, __HIP_MEMORY_SCOPE_WORKGROUP);
  }
  __syncthreads();

  // ---------------- Phase E: store exclusively-owned grad_u rows ----------
  for (int t = tid; t < L_ * D_; t += 256) {
    int l = t >> 7, d = t & (D_ - 1);
    outU[(size_t)(b * L_ + l) * D_ + d] = gradU[l * D_ + d];
  }
}

extern "C" void kernel_launch(void* const* d_in, const int* in_sizes, int n_in,
                              void* d_out, int out_size, void* d_ws, size_t ws_size,
                              hipStream_t stream) {
  (void)in_sizes; (void)n_in; (void)d_ws; (void)ws_size;
  const float* uW    = (const float*)d_in[0];
  const float* vW    = (const float*)d_in[1];
  const int*   nodes = (const int*)d_in[2];
  // d_in[3] = index_emb_posu, d_in[4] = index_emb_posv: band structure is
  // regenerated analytically, so these are unused.
  const int*   negu  = (const int*)d_in[5];
  const int*   negv  = (const int*)d_in[6];

  float* outU = (float*)d_out;
  float* outV = outU + (size_t)BL_ * D_;

  // Outputs are accumulated with atomics -> zero them first (harness poisons).
  zero_out_kernel<<<1024, 256, 0, stream>>>((float*)d_out, out_size);
  skipgram_grad_kernel<<<B_, 256, 0, stream>>>(uW, vW, nodes, negu, negv, outU, outV);
}
